// DSQN_88356067213756
// MI455X (gfx1250) — matrix operand order repaired and checked by reading, not compiled, not run on hardware
//
#include <hip/hip_runtime.h>
#include <hip/hip_bf16.h>

// SNN LIF forward for gfx1250 (MI455X), v2.
// Parallelization: 16 WGs x 256 threads (8 waves). Each WG owns 32 batch rows
// (2 WMMA M-tiles); each wave owns (batch-tile bt, hidden-slice sl) with
// sl covering 32 hidden columns (2 N-tiles). The T=2048 recurrence runs as a
// 2-barrier/step software pipeline; spikes are exchanged through shared LDS
// tiles. All weight B-fragments are register-resident. The per-step x tile
// (32 rows x 64 f32, batch stride T*64) is DMA'd by the Tensor Data Mover
// into a double-buffered LDS tile (wave 0 issues, s_wait_tensorcnt before
// barrier2), fully overlapped with compute.

typedef _Float16 v16h __attribute__((ext_vector_type(16)));
typedef _Float16 v8h  __attribute__((ext_vector_type(8)));
typedef float    v8f  __attribute__((ext_vector_type(8)));

#define TT   2048
#define BBAT 512
#define NOBS 64
#define HID  128
#define NACT 18

#if defined(__HIP_DEVICE_COMPILE__) && __has_builtin(__builtin_amdgcn_tensor_load_to_lds)
#define KUSE_TDM 1
#else
#define KUSE_TDM 0
#endif

#if KUSE_TDM
typedef unsigned tdm_v4u __attribute__((ext_vector_type(4)));
typedef int      tdm_v8i __attribute__((ext_vector_type(8)));
typedef int      tdm_v4i __attribute__((ext_vector_type(4)));

// Issue a TDM 2D tile load: 32 rows x 64 f32, row stride TT*NOBS elements.
// D# packing per CDNA5 ISA ch. 8 (group0: count/lds/global/type,
// group1: data_size, tensor dims, tile dims, dim0 stride).
__device__ __forceinline__ void tdm_load_x(const float* gsrc, unsigned ldsOff) {
  unsigned long long ga = (unsigned long long)(uintptr_t)gsrc;
  tdm_v4u g0;
  g0[0] = 1u;                                        // count=1 (valid user D#)
  g0[1] = ldsOff;                                    // lds_addr (bytes)
  g0[2] = (unsigned)(ga & 0xffffffffu);              // global_addr[31:0]
  g0[3] = (unsigned)((ga >> 32) & 0x01ffffffu)       // global_addr[56:32]
        | 0x80000000u;                               // type=2 ("image")
  tdm_v8i g1;
  g1[0] = 0x20000;                                   // data_size=2 -> 4 bytes
  g1[1] = (int)(64u << 16);                          // tensor_dim0 = 64
  g1[2] = (int)(32u << 16);                          // tensor_dim1 = 32
  g1[3] = (int)(64u << 16);                          // tile_dim0 = 64
  g1[4] = 32;                                        // tile_dim1 = 32
  g1[5] = (int)(TT * NOBS);                          // tensor_dim0_stride
  g1[6] = 0;
  g1[7] = 0;
  tdm_v4i z4 = {0, 0, 0, 0};
#if __clang_major__ >= 23
  tdm_v8i z8 = {0, 0, 0, 0, 0, 0, 0, 0};
  __builtin_amdgcn_tensor_load_to_lds(g0, g1, z4, z4, z8, 0);
#else
  __builtin_amdgcn_tensor_load_to_lds(g0, g1, z4, z4, 0);
#endif
}
#endif // KUSE_TDM

// Build a 16x32 f16 A-fragment row chunk from 32 consecutive floats,
// per ISA 16-bit A layout: halves 0..7 = K[koff..koff+7],
// halves 8..15 = K[16+koff..23+koff] (caller passes p = row + kt*32 + koff).
__device__ __forceinline__ v16h build_a_frag(const float* p) {
  const float4 f0 = *(const float4*)(p);
  const float4 f1 = *(const float4*)(p + 4);
  const float4 f2 = *(const float4*)(p + 16);
  const float4 f3 = *(const float4*)(p + 20);
  v16h a;
  a[0]  = (_Float16)f0.x; a[1]  = (_Float16)f0.y;
  a[2]  = (_Float16)f0.z; a[3]  = (_Float16)f0.w;
  a[4]  = (_Float16)f1.x; a[5]  = (_Float16)f1.y;
  a[6]  = (_Float16)f1.z; a[7]  = (_Float16)f1.w;
  a[8]  = (_Float16)f2.x; a[9]  = (_Float16)f2.y;
  a[10] = (_Float16)f2.z; a[11] = (_Float16)f2.w;
  a[12] = (_Float16)f3.x; a[13] = (_Float16)f3.y;
  a[14] = (_Float16)f3.z; a[15] = (_Float16)f3.w;
  return a;
}

__global__ __launch_bounds__(256, 1)
void snn_lif_wmma(const float* __restrict__ X,   // (B,T,64)
                  const float* __restrict__ HS,  // (B,1,128,2)
                  const float* __restrict__ W1,  // (128,64)
                  const float* __restrict__ B1,  // (128)
                  const float* __restrict__ BE1, // (128)
                  const float* __restrict__ W2,  // (128,128)
                  const float* __restrict__ B2,  // (128)
                  const float* __restrict__ BE2, // (128)
                  const float* __restrict__ W3,  // (18,128)
                  const float* __restrict__ B3,  // (18)
                  float* __restrict__ OUT)       // (B,T,18)
{
  __shared__ __align__(64) float    xbuf[2][32 * NOBS];   // 16 KB double buffer
  __shared__ __align__(64) _Float16 spk1T[2][16 * HID];   // 8 KB (per bt tile)
  __shared__ __align__(64) _Float16 spk2T[2][16 * HID];   // 8 KB

  const int tid    = threadIdx.x;
  const int lane   = tid & 31;
  const int wave   = tid >> 5;
  const int lane16 = lane & 15;
  const int hi     = (lane & 16) ? 1 : 0;
  const int koff   = hi * 8;   // A-fragment per-lane K sub-offset
  const int rowOff = hi * 8;   // C/D-fragment row offset
  const int bt     = wave >> 2;        // batch tile within WG (0..1)
  const int sl     = wave & 3;         // hidden-column slice (0..3)
  const int n0     = sl * 2;           // first of this wave's 2 N-tiles
  const int b0wg   = blockIdx.x * 32;  // WG batch base
  const int bMine  = b0wg + bt * 16;   // this wave's batch tile base

#if KUSE_TDM
  // Kick off the t=0 observation tile while we set up weights.
  if (wave == 0)
    tdm_load_x(X + (size_t)b0wg * TT * NOBS,
               (unsigned)(uintptr_t)&xbuf[0][0]);
#endif

  // ---- register-resident B-fragments for this wave's 32-column slice.
  // B layout: lane holds col N = n*16 + lane%16, K = hi*8+0..7 and 16+hi*8+0..7.
  v16h w1f[2][2];  // [kt][local n]
#pragma unroll
  for (int kt = 0; kt < 2; ++kt)
#pragma unroll
    for (int j = 0; j < 2; ++j) {
      const int col = (n0 + j) * 16 + lane16;
      const float* src = W1 + col * NOBS + kt * 32 + hi * 8;
      v16h v;
#pragma unroll
      for (int q = 0; q < 16; ++q) v[q] = (_Float16)src[q < 8 ? q : q + 8];
      w1f[kt][j] = v;
    }

  v16h w2f[4][2];  // [kt][local n]
#pragma unroll
  for (int kt = 0; kt < 4; ++kt)
#pragma unroll
    for (int j = 0; j < 2; ++j) {
      const int col = (n0 + j) * 16 + lane16;
      const float* src = W2 + col * HID + kt * 32 + hi * 8;
      v16h v;
#pragma unroll
      for (int q = 0; q < 16; ++q) v[q] = (_Float16)src[q < 8 ? q : q + 8];
      w2f[kt][j] = v;
    }

  // W3 fragments: only slices 0/1 use them (N-tile = sl); zero-pad cols >= 18.
  v16h w3f[4];
#pragma unroll
  for (int kt = 0; kt < 4; ++kt) {
    const int col = sl * 16 + lane16;
    v16h v;
    if (sl < 2 && col < NACT) {
      const float* src = W3 + col * HID + kt * 32 + hi * 8;
#pragma unroll
      for (int q = 0; q < 16; ++q) v[q] = (_Float16)src[q < 8 ? q : q + 8];
    } else {
#pragma unroll
      for (int q = 0; q < 16; ++q) v[q] = (_Float16)0.0f;
    }
    w3f[kt] = v;
  }

  // ---- per-lane bias / clipped beta for this slice (column-indexed)
  float b1v[2], be1v[2], b2v[2], be2v[2];
#pragma unroll
  for (int j = 0; j < 2; ++j) {
    const int col = (n0 + j) * 16 + lane16;
    b1v[j]  = B1[col];
    be1v[j] = fminf(fmaxf(BE1[col], 0.0f), 1.0f);
    b2v[j]  = B2[col];
    be2v[j] = fminf(fmaxf(BE2[col], 0.0f), 1.0f);
  }
  const int outCol = sl * 16 + lane16;
  const float b3v = (sl < 2 && outCol < NACT) ? B3[outCol] : 0.0f;

  // ---- membrane state (C/D layout: lane holds one col, 8 rows per N-tile)
  float mem1[16], mem2[16];
#pragma unroll
  for (int j = 0; j < 2; ++j)
#pragma unroll
    for (int r = 0; r < 8; ++r) {
      const int b = bMine + r + rowOff;
      const int h = (n0 + j) * 16 + lane16;
      mem1[j * 8 + r] = HS[((size_t)b * HID + h) * 2 + 0];
      mem2[j * 8 + r] = HS[((size_t)b * HID + h) * 2 + 1];
    }

  _Float16* spk1    = &spk1T[bt][0];
  _Float16* spk2    = &spk2T[bt][0];
  _Float16* spk1Row = spk1 + lane16 * HID;
  _Float16* spk2Row = spk2 + lane16 * HID;

#if KUSE_TDM
  if (wave == 0) __builtin_amdgcn_s_wait_tensorcnt(0);  // t=0 tile landed
#endif
  __syncthreads();

  for (int t = 0; t < TT; ++t) {
    // ================= phase P0: GEMM1 + LIF1 =================
#if KUSE_TDM
    if (wave == 0 && t + 1 < TT)   // prefetch next step's x tile via TDM
      tdm_load_x(X + ((size_t)b0wg * TT + (t + 1)) * NOBS,
                 (unsigned)(uintptr_t)&xbuf[(t + 1) & 1][0]);
    const float* xrow = &xbuf[t & 1][(bt * 16 + lane16) * NOBS];
#else
    const float* xrow = X + ((size_t)(bMine + lane16) * TT + t) * NOBS;
#endif

    v16h ax[2];
#pragma unroll
    for (int kt = 0; kt < 2; ++kt)
      ax[kt] = build_a_frag(xrow + kt * 32 + koff);

    v8f acc[2];
#pragma unroll
    for (int j = 0; j < 2; ++j) {
      v8f c = {};
      c = __builtin_amdgcn_wmma_f32_16x16x32_f16(false, ax[0], false,
                                                 w1f[0][j], (short)0, c,
                                                 false, false);
      c = __builtin_amdgcn_wmma_f32_16x16x32_f16(false, ax[1], false,
                                                 w1f[1][j], (short)0, c,
                                                 false, false);
      acc[j] = c;
    }

#pragma unroll
    for (int j = 0; j < 2; ++j) {
      const int col = (n0 + j) * 16 + lane16;
#pragma unroll
      for (int r = 0; r < 8; ++r) {
        float m = mem1[j * 8 + r];
        const float reset = (m > 1.0f) ? 1.0f : 0.0f;
        m = be1v[j] * m + (acc[j][r] + b1v[j]) - reset;
        mem1[j * 8 + r] = m;
        spk1[(r + rowOff) * HID + col] =
            (m > 1.0f) ? (_Float16)1.0f : (_Float16)0.0f;
      }
    }
    __syncthreads();  // barrier1: spk1 complete across all slices

    // ================= phase P1: GEMM2 + LIF2 =================
    v16h sA[4];
#pragma unroll
    for (int kt = 0; kt < 4; ++kt) {
      const v8h c0 = *(const v8h*)(spk1Row + kt * 32 + koff);
      const v8h c1 = *(const v8h*)(spk1Row + kt * 32 + 16 + koff);
      sA[kt] = __builtin_shufflevector(c0, c1, 0, 1, 2, 3, 4, 5, 6, 7,
                                       8, 9, 10, 11, 12, 13, 14, 15);
    }

#pragma unroll
    for (int j = 0; j < 2; ++j) {
      v8f c = {};
#pragma unroll
      for (int kt = 0; kt < 4; ++kt)
        c = __builtin_amdgcn_wmma_f32_16x16x32_f16(false, sA[kt], false,
                                                   w2f[kt][j], (short)0, c,
                                                   false, false);
      acc[j] = c;
    }

#pragma unroll
    for (int j = 0; j < 2; ++j) {
      const int col = (n0 + j) * 16 + lane16;
#pragma unroll
      for (int r = 0; r < 8; ++r) {
        float m = mem2[j * 8 + r];
        const float reset = (m > 1.0f) ? 1.0f : 0.0f;
        m = be2v[j] * m + (acc[j][r] + b2v[j]) - reset;
        mem2[j * 8 + r] = m;
        spk2[(r + rowOff) * HID + col] =
            (m > 1.0f) ? (_Float16)1.0f : (_Float16)0.0f;
      }
    }

#if KUSE_TDM
    if (wave == 0 && t + 1 < TT)
      __builtin_amdgcn_s_wait_tensorcnt(0);  // x[t+1] resident before barrier2
#endif
    __syncthreads();  // barrier2: spk2 complete (and x[t+1] staged)

    // ================= phase P2: GEMM3 + output (slices 0/1 only) =========
    if (sl < 2) {
      v16h s2[4];
#pragma unroll
      for (int kt = 0; kt < 4; ++kt) {
        const v8h c0 = *(const v8h*)(spk2Row + kt * 32 + koff);
        const v8h c1 = *(const v8h*)(spk2Row + kt * 32 + 16 + koff);
        s2[kt] = __builtin_shufflevector(c0, c1, 0, 1, 2, 3, 4, 5, 6, 7,
                                         8, 9, 10, 11, 12, 13, 14, 15);
      }
      v8f c = {};
#pragma unroll
      for (int kt = 0; kt < 4; ++kt)
        c = __builtin_amdgcn_wmma_f32_16x16x32_f16(false, s2[kt], false,
                                                   w3f[kt], (short)0, c,
                                                   false, false);
      if (outCol < NACT) {
#pragma unroll
        for (int r = 0; r < 8; ++r) {
          const size_t b = (size_t)(bMine + r + rowOff);
          OUT[(b * TT + t) * NACT + outCol] = c[r] + b3v;
        }
      }
    }
  }
}

extern "C" void kernel_launch(void* const* d_in, const int* in_sizes, int n_in,
                              void* d_out, int out_size, void* d_ws, size_t ws_size,
                              hipStream_t stream) {
  const float* X   = (const float*)d_in[0]; // state_batch (512,2048,64)
  const float* HS  = (const float*)d_in[1]; // hidden_states (512,1,128,2)
  const float* W1  = (const float*)d_in[2];
  const float* B1  = (const float*)d_in[3];
  const float* BE1 = (const float*)d_in[4];
  const float* W2  = (const float*)d_in[5];
  const float* B2  = (const float*)d_in[6];
  const float* BE2 = (const float*)d_in[7];
  const float* W3  = (const float*)d_in[8];
  const float* B3  = (const float*)d_in[9];
  float* OUT = (float*)d_out;

  // 16 WGs x 8 waves: each WG owns 32 batch rows, each wave a
  // (batch-tile, 32-hidden-col slice). 128 persistent waves total.
  snn_lif_wmma<<<dim3(BBAT / 32), dim3(256), 0, stream>>>(
      X, HS, W1, B1, BE1, W2, B2, BE2, W3, B3, OUT);
}
